// Motion_80934363726021
// MI455X (gfx1250) — matrix operand order repaired and checked by reading, not compiled, not run on hardware
//
#include <hip/hip_runtime.h>
#include <hip/hip_bf16.h>
#include <math.h>

// ---------------------------------------------------------------------------
// CDNA5 (gfx1250) wave32 WMMA implementation of the PointLSTM forward pass.
// All einsum GEMMs run on v_wmma_f32_16x16x32_bf16 with operands pre-packed
// into per-lane fragments, contiguous k order (k = 16*h + e).
// ---------------------------------------------------------------------------

typedef __attribute__((ext_vector_type(16))) __bf16 v16bf;
typedef __attribute__((ext_vector_type(8)))  float  v8f;

#define BB   2
#define TT   32
#define NN1  128
#define KK1  16
#define RSBN 0.99999500003749969f   // 1/sqrt(1+1e-5)

__device__ __forceinline__ unsigned short f2bf(float f){
  unsigned int u = __float_as_uint(f);
  u += 0x7fffu + ((u >> 16) & 1u);          // round-to-nearest-even
  return (unsigned short)(u >> 16);
}

// B-matrix (K x 16) fragment offset: lane = (k%32)/16*16 + col%16, elem = k%16
__device__ __forceinline__ size_t bpack_off(int kIdx, int col, int KB){
  int kb = kIdx >> 5, kc = kIdx & 31;
  int h  = kc >> 4,  e  = kc & 15;
  int nt = col >> 4, n16 = col & 15;
  return ((size_t)(nt * KB + kb) * 32 + (h * 16 + n16)) * 16 + e;
}
// A-matrix (16 x K) fragment offset, contiguous k order matching B (K = 16*h + e)
__device__ __forceinline__ size_t apack_off(int o, int c, int KB){
  int mt = o >> 4, m16 = o & 15, kb = c >> 5, kc = c & 31;
  int h = kc >> 4;
  int e = kc & 15;
  return ((size_t)(mt * KB + kb) * 32 + (h * 16 + m16)) * 16 + e;
}
__device__ __forceinline__ float sigm(float x){ return 1.f / (1.f + __expf(-x)); }

// ---------------------------------------------------------------------------
// utility kernels
// ---------------------------------------------------------------------------
__global__ void zero_u16_kernel(unsigned short* p, long n){
  long i = (long)blockIdx.x * blockDim.x + threadIdx.x;
  if (i < n) p[i] = 0;
}
__global__ void zero_f32_kernel(float* p, long n){
  long i = (long)blockIdx.x * blockDim.x + threadIdx.x;
  if (i < n) p[i] = 0.f;
}

// input (2,32,256,4) -> x[B][4][T][128] with N-stride 2
__global__ void init_x_kernel(const float* __restrict__ in, float* __restrict__ x){
  long i = (long)blockIdx.x * blockDim.x + threadIdx.x;
  long total = (long)BB * 4 * TT * NN1;
  if (i >= total) return;
  int n = (int)(i % NN1); long r = i / NN1;
  int t = (int)(r % TT);  r /= TT;
  int c = (int)(r % 4);
  int b = (int)(r / 4);
  x[i] = in[(((long)b * TT + t) * 256 + 2 * n) * 4 + c];
}

// fea[b][c][t][n] (c<4) = x[b][c][t][n*stride]
__global__ void copy_pos_kernel(const float* __restrict__ x, float* __restrict__ fea,
                                int Ctot, int nd, int stride){
  long i = (long)blockIdx.x * blockDim.x + threadIdx.x;
  long total = (long)BB * 4 * TT * nd;
  if (i >= total) return;
  int n = (int)(i % nd); long r = i / nd;
  int t = (int)(r % TT); r /= TT;
  int c = (int)(r % 4);
  int b = (int)(r / 4);
  fea[(((long)b * Ctot + c) * TT + t) * nd + n] =
      x[(((long)b * 4 + c) * TT + t) * NN1 + (long)n * stride];
}

// ---------------------------------------------------------------------------
// KNN: per-batch block, refs staged in LDS, lexicographic (d2, idx) k-pass
// selection matching jax top_k tie-breaking.
// ---------------------------------------------------------------------------
__global__ void knn_kernel(const float* __restrict__ q, long qB, long qC, int qs, int Nq,
                           const float* __restrict__ r, long rB, long rC, int Nr,
                           int* __restrict__ idx, int k, long idxB){
  __shared__ float rp[3][NN1];
  int b = blockIdx.x;
  int tid = threadIdx.x;
  for (int j = tid; j < Nr; j += blockDim.x){
    rp[0][j] = r[b * rB + 0 * rC + j];
    rp[1][j] = r[b * rB + 1 * rC + j];
    rp[2][j] = r[b * rB + 2 * rC + j];
  }
  __syncthreads();
  if (tid >= Nq) return;
  float qx = q[b * qB + 0 * qC + (long)tid * qs];
  float qy = q[b * qB + 1 * qC + (long)tid * qs];
  float qz = q[b * qB + 2 * qC + (long)tid * qs];
  float pd = -1.f; int pi = -1;
  for (int j = 0; j < k; ++j){
    float bd = 3.4e38f; int bi = 0x7fffffff;
    for (int m = 0; m < Nr; ++m){
      float dx = qx - rp[0][m], dy = qy - rp[1][m], dz = qz - rp[2][m];
      float d2 = dx * dx + dy * dy + dz * dz;
      bool gt = (d2 > pd) || (d2 == pd && m > pi);
      bool lt = (d2 < bd) || (d2 == bd && m < bi);
      if (gt && lt){ bd = d2; bi = m; }
    }
    idx[b * idxB + (long)tid * k + j] = bi;
    pd = bd; pi = bi;
  }
}

// ---------------------------------------------------------------------------
// operand packing kernels (f32 -> bf16 fragment-order scatter)
// ---------------------------------------------------------------------------
__global__ void pack_w_kernel(const float* __restrict__ W, unsigned short* __restrict__ Ap,
                              int Co, int Ci, int KB){
  long i = (long)blockIdx.x * blockDim.x + threadIdx.x;
  if (i >= (long)Co * Ci) return;
  int o = (int)(i / Ci), c = (int)(i % Ci);
  Ap[apack_off(o, c, KB)] = f2bf(W[i]);
}

// stage1 ret1: c<3 -> gathered - center ; c==3 -> gathered
__global__ void pack_stage1_kernel(const float* __restrict__ x, const int* __restrict__ idx,
                                   unsigned short* __restrict__ Bp, int t){
  long i = (long)blockIdx.x * blockDim.x + threadIdx.x;
  long total = (long)BB * 4 * NN1 * KK1;
  if (i >= total) return;
  int kk = (int)(i % KK1); long r = i / KK1;
  int n  = (int)(r % NN1); r /= NN1;
  int c  = (int)(r % 4);
  int b  = (int)(r / 4);
  int g  = idx[((long)b * NN1 + n) * KK1 + kk];
  const float* xb = x + (((long)b * 4 + c) * TT + t) * NN1;
  float v = xb[g];
  if (c < 3) v -= xb[n];
  int col = (b * NN1 + n) * KK1 + kk;
  Bp[bpack_off(c, col, 1)] = f2bf(v);
}

// LSTM comb = [ x (C ch, broadcast over k) ; h gathered (4+hd ch, first 4 -= pos) ]
__global__ void pack_lstm_kernel(const float* __restrict__ fea, const float* __restrict__ h,
                                 const int* __restrict__ idx, unsigned short* __restrict__ Bp,
                                 int C, int N, int k, int t, int KB){
  int Ci = 2 * C;   // C == 4 + hd for every LSTM in this net
  long i = (long)blockIdx.x * blockDim.x + threadIdx.x;
  long total = (long)BB * Ci * N * k;
  if (i >= total) return;
  int kk = (int)(i % k); long r = i / k;
  int n  = (int)(r % N); r /= N;
  int c  = (int)(r % Ci);
  int b  = (int)(r / Ci);
  float v;
  if (c < C){
    v = fea[(((long)b * C + c) * TT + t) * N + n];
  } else {
    int ch = c - C;
    int g  = idx[((long)b * N + n) * k + kk];
    v = h[((long)b * C + ch) * N + g];
    if (ch < 4) v -= fea[(((long)b * C + ch) * TT + t) * N + n];
  }
  int col = (b * N + n) * k + kk;
  Bp[bpack_off(c, col, KB)] = f2bf(v);
}

__global__ void gather_c_kernel(const float* __restrict__ cbuf, const int* __restrict__ idx,
                                float* __restrict__ cg, int hd, int N, int k){
  long i = (long)blockIdx.x * blockDim.x + threadIdx.x;
  long total = (long)BB * hd * N * k;
  if (i >= total) return;
  int kk = (int)(i % k); long r = i / k;
  int n  = (int)(r % N); r /= N;
  int ch = (int)(r % hd);
  int b  = (int)(r / hd);
  int g  = idx[((long)b * N + n) * k + kk];
  long Mcols = (long)BB * N * k;
  cg[(long)ch * Mcols + ((long)b * N + n) * k + kk] = cbuf[((long)b * hd + ch) * N + g];
}

// stage grouping: fea half = [ gathered(prev) hd ; center(cur) hd ]
__global__ void pack_stage_fea_kernel(const float* __restrict__ fl, const int* __restrict__ idx,
                                      unsigned short* __restrict__ Bp,
                                      int C, int hd, int Nf, int nd, int k,
                                      int t, int tp, int s, int KB){
  long i = (long)blockIdx.x * blockDim.x + threadIdx.x;
  long total = (long)BB * 2 * hd * nd * k;
  if (i >= total) return;
  int kk = (int)(i % k); long r = i / k;
  int n  = (int)(r % nd); r /= nd;
  int c2 = (int)(r % (2 * hd));
  int b  = (int)(r / (2 * hd));
  float v;
  if (c2 < hd){
    int g = idx[((long)b * nd + n) * k + kk];
    v = fl[(((long)b * C + 4 + c2) * TT + tp) * Nf + g];
  } else {
    v = fl[(((long)b * C + 4 + (c2 - hd)) * TT + t) * Nf + (long)n * s];
  }
  int col = (b * nd + n) * k + kk;
  Bp[bpack_off(c2, col, KB)] = f2bf(v);
}

// stage grouping: mot half = gathered(prev)[0..3] - center(cur)[0..3]
__global__ void pack_stage_mot_kernel(const float* __restrict__ fl, const int* __restrict__ idx,
                                      unsigned short* __restrict__ Bp,
                                      int C, int Nf, int nd, int k, int t, int tp, int s){
  long i = (long)blockIdx.x * blockDim.x + threadIdx.x;
  long total = (long)BB * 4 * nd * k;
  if (i >= total) return;
  int kk = (int)(i % k); long r = i / k;
  int n  = (int)(r % nd); r /= nd;
  int c  = (int)(r % 4);
  int b  = (int)(r / 4);
  int g  = idx[((long)b * nd + n) * k + kk];
  float v = fl[(((long)b * C + c) * TT + tp) * Nf + g]
          - fl[(((long)b * C + c) * TT + t) * Nf + (long)n * s];
  int col = (b * nd + n) * k + kk;
  Bp[bpack_off(c, col, 1)] = f2bf(v);
}

__global__ void pack_plain_kernel(const float* __restrict__ src, unsigned short* __restrict__ Bp,
                                  int C, int TN, int KB){
  long i = (long)blockIdx.x * blockDim.x + threadIdx.x;
  long total = (long)BB * C * TN;
  if (i >= total) return;
  int m = (int)(i % TN); long r = i / TN;
  int c = (int)(r % C);
  int b = (int)(r / C);
  int col = b * TN + m;
  Bp[bpack_off(c, col, KB)] = f2bf(src[((long)b * C + c) * TN + m]);
}

// ---------------------------------------------------------------------------
// WMMA GEMM: D[Co x cols] = W x X, fragments pre-packed. MODE bits:
//   1 = +bias   2 = BN(g*x*rs+b)+ReLU   4 = accumulate into outF   8 = bf16 pack out
// ---------------------------------------------------------------------------
template<int MODE>
__global__ void gemm_wmma_kernel(const unsigned short* __restrict__ Ap,
                                 const unsigned short* __restrict__ Bp,
                                 int KB, int Mtiles, int Ntiles,
                                 float* __restrict__ outF,
                                 unsigned short* __restrict__ outP, int outKB, int Mcols,
                                 const float* __restrict__ bias,
                                 const float* __restrict__ gamma,
                                 const float* __restrict__ beta){
  int wv = (blockIdx.x * blockDim.x + threadIdx.x) >> 5;   // wave-uniform
  if (wv >= Mtiles * Ntiles) return;
  int lane = threadIdx.x & 31;
  int mt = wv / Ntiles, nt = wv - mt * Ntiles;
  const unsigned short* a0 = Ap + ((size_t)mt * KB * 32 + lane) * 16;
  const unsigned short* b0 = Bp + ((size_t)nt * KB * 32 + lane) * 16;
  v8f acc = {};
  for (int kb = 0; kb < KB; ++kb){
    v16bf av = *(const v16bf*)(a0 + (size_t)kb * 512);
    v16bf bv = *(const v16bf*)(b0 + (size_t)kb * 512);
    acc = __builtin_amdgcn_wmma_f32_16x16x32_bf16(false, av, false, bv,
                                                  (short)0, acc, false, false);
  }
  int h = lane >> 4;
  int col = nt * 16 + (lane & 15);
  #pragma unroll
  for (int v = 0; v < 8; ++v){
    int o = mt * 16 + v + 8 * h;
    float y = acc[v];
    if (MODE & 1) y += bias[o];
    if (MODE & 2){ y = gamma[o] * y * RSBN + beta[o]; y = fmaxf(y, 0.f); }
    if (MODE & 8)      outP[bpack_off(o, col, outKB)] = f2bf(y);
    else if (MODE & 4) outF[(size_t)o * Mcols + col] += y;
    else               outF[(size_t)o * Mcols + col]  = y;
  }
}

// ---------------------------------------------------------------------------
// reductions / epilogues
// ---------------------------------------------------------------------------
__global__ void maxk_kernel(const float* __restrict__ in, float* __restrict__ out,
                            int Co, int nd, int k, int t, int Ctot, int chOff, int Mcols){
  long i = (long)blockIdx.x * blockDim.x + threadIdx.x;
  long total = (long)BB * Co * nd;
  if (i >= total) return;
  int n  = (int)(i % nd); long r = i / nd;
  int ch = (int)(r % Co);
  int b  = (int)(r / Co);
  long base = (long)ch * Mcols + ((long)b * nd + n) * k;
  float m = -3.4e38f;
  for (int kk = 0; kk < k; ++kk) m = fmaxf(m, in[base + kk]);
  out[(((long)b * Ctot + chOff + ch) * TT + t) * nd + n] = m;
}

__global__ void lstm_act_kernel(const float* __restrict__ gates, const float* __restrict__ cg,
                                const float* __restrict__ fea, float* __restrict__ h,
                                float* __restrict__ cbuf, float* __restrict__ fl,
                                int C, int hd, int N, int k, int t){
  long i = (long)blockIdx.x * blockDim.x + threadIdx.x;
  long total = (long)BB * C * N;
  if (i >= total) return;
  int n  = (int)(i % N); long r = i / N;
  int ch = (int)(r % C);
  int b  = (int)(r / C);
  if (ch < 4){
    float p = fea[(((long)b * C + ch) * TT + t) * N + n];
    h[((long)b * C + ch) * N + n] = p;
    fl[(((long)b * C + ch) * TT + t) * N + n] = p;
    return;
  }
  int cc = ch - 4;
  long Mcols = (long)BB * N * k;
  long base  = ((long)b * N + n) * k;
  float cm = -3.4e38f, hm = -3.4e38f;
  for (int kk = 0; kk < k; ++kk){
    long col = base + kk;
    float gi = gates[(long)cc * Mcols + col];
    float gf = gates[((long)hd + cc) * Mcols + col];
    float go = gates[((long)2 * hd + cc) * Mcols + col];
    float gg = gates[((long)3 * hd + cc) * Mcols + col];
    float cn = cg[(long)cc * Mcols + col];
    float cnk = sigm(gf) * cn + sigm(gi) * tanhf(gg);
    float hnk = sigm(go) * tanhf(cnk);
    cm = fmaxf(cm, cnk); hm = fmaxf(hm, hnk);
  }
  cbuf[((long)b * hd + cc) * N + n] = cm;
  h[((long)b * C + ch) * N + n] = hm;
  fl[(((long)b * C + ch) * TT + t) * N + n] = hm;
}

__global__ void pool_stage5_kernel(const float* __restrict__ g, float* __restrict__ oe,
                                   int Mcols, int TN){
  long i = (long)blockIdx.x * blockDim.x + threadIdx.x;
  if (i >= (long)BB * 1024) return;
  int ch = (int)(i % 1024);
  int b  = (int)(i / 1024);
  float m = -3.4e38f;
  for (int j = 0; j < TN; ++j) m = fmaxf(m, g[(long)ch * Mcols + (long)b * TN + j]);
  oe[(long)b * 1472 + ch] = m;
}

__global__ void pool_end_kernel(const float* __restrict__ fea, const float* __restrict__ fl,
                                float* __restrict__ oe, int C, int TN, int off, int hd){
  long i = (long)blockIdx.x * blockDim.x + threadIdx.x;
  if (i >= (long)BB * hd) return;
  int ch = (int)(i % hd);
  int b  = (int)(i / hd);
  long base = ((long)b * C + 4 + ch) * TN;
  float m = -3.4e38f;
  for (int j = 0; j < TN; ++j) m = fmaxf(m, fea[base + j] + fl[base + j]);
  oe[(long)b * 1472 + off + ch] = m;
}

// fused global-BN + stage6 linear head
__global__ void head_kernel(const float* __restrict__ oe, const float* __restrict__ W,
                            const float* __restrict__ bias, const float* __restrict__ gg,
                            const float* __restrict__ gb, float* __restrict__ out){
  int i = blockIdx.x * blockDim.x + threadIdx.x;
  if (i >= BB * 28) return;
  int o = i % 28, b = i / 28;
  float acc = bias[o];
  for (int c = 0; c < 1472; ++c)
    acc += W[o * 1472 + c] * (gg[c] * oe[(long)b * 1472 + c] * RSBN + gb[c]);
  out[i] = acc;
}

// ---------------------------------------------------------------------------
// host side
// ---------------------------------------------------------------------------
static inline int gblocks(long n){ return (int)((n + 255) / 256); }

static void launch_gemm(hipStream_t s, int mode,
                        const unsigned short* Ap, const unsigned short* Bp,
                        int KB, int Co, int cols,
                        float* outF, unsigned short* outP, int outKB,
                        const float* bias, const float* gamma, const float* beta){
  int Mt = Co / 16, Nt = cols / 16;
  dim3 grid((Mt * Nt + 3) / 4), blk(128);
  switch (mode){
    case 1:  gemm_wmma_kernel<1 ><<<grid, blk, 0, s>>>(Ap, Bp, KB, Mt, Nt, outF, outP, outKB, cols, bias, gamma, beta); break;
    case 3:  gemm_wmma_kernel<3 ><<<grid, blk, 0, s>>>(Ap, Bp, KB, Mt, Nt, outF, outP, outKB, cols, bias, gamma, beta); break;
    case 7:  gemm_wmma_kernel<7 ><<<grid, blk, 0, s>>>(Ap, Bp, KB, Mt, Nt, outF, outP, outKB, cols, bias, gamma, beta); break;
    case 11: gemm_wmma_kernel<11><<<grid, blk, 0, s>>>(Ap, Bp, KB, Mt, Nt, outF, outP, outKB, cols, bias, gamma, beta); break;
  }
}

extern "C" void kernel_launch(void* const* d_in, const int* in_sizes, int n_in,
                              void* d_out, int out_size, void* d_ws, size_t ws_size,
                              hipStream_t stream){
  (void)in_sizes; (void)n_in; (void)out_size; (void)ws_size;
  // ---- parameter table (setup_inputs flatten order) ----
  const float* IN   = (const float*)d_in[0];
  const float* W1A  = (const float*)d_in[1];  const float* B1A = (const float*)d_in[2];
  const float* G1A  = (const float*)d_in[3];  const float* E1A = (const float*)d_in[4];
  const float* W1B  = (const float*)d_in[5];  const float* B1B = (const float*)d_in[6];
  const float* G1B  = (const float*)d_in[7];  const float* E1B = (const float*)d_in[8];
  const float* WL1  = (const float*)d_in[9];  const float* BL1 = (const float*)d_in[10];
  const float* W2F  = (const float*)d_in[11]; const float* B2F = (const float*)d_in[12];
  const float* G2F  = (const float*)d_in[13]; const float* E2F = (const float*)d_in[14];
  const float* W2M  = (const float*)d_in[15]; const float* B2M = (const float*)d_in[16];
  const float* G2M  = (const float*)d_in[17]; const float* E2M = (const float*)d_in[18];
  const float* WL2  = (const float*)d_in[19]; const float* BL2 = (const float*)d_in[20];
  const float* W3F  = (const float*)d_in[21]; const float* B3F = (const float*)d_in[22];
  const float* G3F  = (const float*)d_in[23]; const float* E3F = (const float*)d_in[24];
  const float* W3M  = (const float*)d_in[25]; const float* B3M = (const float*)d_in[26];
  const float* G3M  = (const float*)d_in[27]; const float* E3M = (const float*)d_in[28];
  const float* WL3  = (const float*)d_in[29]; const float* BL3 = (const float*)d_in[30];
  const float* W4F  = (const float*)d_in[31]; const float* B4F = (const float*)d_in[32];
  const float* G4F  = (const float*)d_in[33]; const float* E4F = (const float*)d_in[34];
  const float* W4M  = (const float*)d_in[35]; const float* B4M = (const float*)d_in[36];
  const float* G4M  = (const float*)d_in[37]; const float* E4M = (const float*)d_in[38];
  const float* W5   = (const float*)d_in[39]; const float* B5  = (const float*)d_in[40];
  const float* G5   = (const float*)d_in[41]; const float* E5  = (const float*)d_in[42];
  const float* W6   = (const float*)d_in[43]; const float* B6  = (const float*)d_in[44];
  const float* GBG  = (const float*)d_in[45]; const float* GBB = (const float*)d_in[46];
  float* OUT = (float*)d_out;

  // ---- workspace bump allocator ----
  char* w = (char*)d_ws; size_t off = 0;
  auto alloc = [&](size_t bytes)->void*{
    off = (off + 255) & ~(size_t)255;
    void* p = w + off; off += bytes; return p;
  };
  auto af = [&](long n){ return (float*)alloc(sizeof(float) * n); };
  auto au = [&](long n){ return (unsigned short*)alloc(2 * n); };

  float* x     = af((long)BB * 4 * TT * NN1);
  float* fea1  = af((long)BB * 68 * TT * NN1);
  float* fea1l = af((long)BB * 68 * TT * NN1);
  float* fea2  = af((long)BB * 132 * TT * 64);
  float* fea2l = af((long)BB * 132 * TT * 64);
  float* fea3  = af((long)BB * 260 * TT * 32);
  float* fea3l = af((long)BB * 260 * TT * 32);
  float* f4    = af((long)BB * 512 * TT * 16);
  int*   idxL  = (int*)alloc(sizeof(int) * (long)BB * 128 * 48);
  float* hbuf  = af((long)BB * 68 * 128);
  float* cbuf  = af((long)BB * 64 * 128);
  float* cg    = af(262144);
  float* arena = af(1048576);              // shared: LSTM gates / stage GEMM outs
  float* oe    = af((long)BB * 1472);

  size_t packStart = (off + 255) & ~(size_t)255;
  unsigned short* pS1a = au(32  * 4096);   unsigned short* pS1m = au(32  * 4096);
  unsigned short* pL1  = au(160 * 4096);
  unsigned short* pS2f = au(128 * 6144);   unsigned short* pS2m = au(32  * 6144);
  unsigned short* pL2  = au(288 * 2048);
  unsigned short* pS3f = au(256 * 3072);   unsigned short* pS3m = au(32  * 3072);
  unsigned short* pL3  = au(544 * 1024);
  unsigned short* pS4f = au(512 * 768);    unsigned short* pS4m = au(32  * 768);
  unsigned short* pS5  = au(512 * 1024);
  unsigned short* wS1a = au(32 * 32);      unsigned short* wS1b = au(64 * 32);
  unsigned short* wL1  = au(256 * 160);
  unsigned short* wS2f = au(128 * 128);    unsigned short* wS2m = au(128 * 32);
  unsigned short* wL2  = au(512 * 288);
  unsigned short* wS3f = au(256 * 256);    unsigned short* wS3m = au(256 * 32);
  unsigned short* wL3  = au(1024 * 544);
  unsigned short* wS4f = au(512 * 512);    unsigned short* wS4m = au(512 * 32);
  unsigned short* wS5  = au(1024 * 512);
  size_t packEnd = off;

  // zero the whole packed region once (zero-pad K rows must read as 0)
  long nz = (long)((packEnd - packStart) / 2);
  zero_u16_kernel<<<gblocks(nz), 256, 0, stream>>>((unsigned short*)(w + packStart), nz);

  // pack weights
  auto packW = [&](const float* W, unsigned short* Ap, int Co, int Ci, int KB){
    pack_w_kernel<<<gblocks((long)Co * Ci), 256, 0, stream>>>(W, Ap, Co, Ci, KB);
  };
  packW(W1A, wS1a, 32, 4, 1);     packW(W1B, wS1b, 64, 32, 1);
  packW(WL1, wL1, 256, 136, 5);
  packW(W2F, wS2f, 128, 128, 4);  packW(W2M, wS2m, 128, 4, 1);
  packW(WL2, wL2, 512, 264, 9);
  packW(W3F, wS3f, 256, 256, 8);  packW(W3M, wS3m, 256, 4, 1);
  packW(WL3, wL3, 1024, 520, 17);
  packW(W4F, wS4f, 512, 512, 16); packW(W4M, wS4m, 512, 4, 1);
  packW(W5,  wS5, 1024, 512, 16);

  // x and position channels
  init_x_kernel<<<gblocks((long)BB * 4 * TT * NN1), 256, 0, stream>>>(IN, x);
  copy_pos_kernel<<<gblocks((long)BB * 4 * TT * 128), 256, 0, stream>>>(x, fea1,  68, 128, 1);
  copy_pos_kernel<<<gblocks((long)BB * 4 * TT * 64 ), 256, 0, stream>>>(x, fea2, 132, 64, 2);
  copy_pos_kernel<<<gblocks((long)BB * 4 * TT * 32 ), 256, 0, stream>>>(x, fea3, 260, 32, 4);

  // ---- stage1: KNN(k=16) -> MLP 4->32->64 (WMMA, chained packed) -> max_k ----
  for (int t = 0; t < TT; ++t){
    knn_kernel<<<BB, 128, 0, stream>>>(x + (long)t * NN1, (long)4 * TT * NN1, (long)TT * NN1, 1, NN1,
                                       x + (long)t * NN1, (long)4 * TT * NN1, (long)TT * NN1, NN1,
                                       idxL, KK1, (long)NN1 * KK1);
    pack_stage1_kernel<<<gblocks((long)BB * 4 * NN1 * KK1), 256, 0, stream>>>(x, idxL, pS1a, t);
    launch_gemm(stream, 11, wS1a, pS1a, 1, 32, 4096, nullptr, pS1m, 1, B1A, G1A, E1A);
    launch_gemm(stream, 3,  wS1b, pS1m, 1, 64, 4096, arena, nullptr, 0, B1B, G1B, E1B);
    maxk_kernel<<<gblocks((long)BB * 64 * NN1), 256, 0, stream>>>(arena, fea1, 64, NN1, KK1, t, 68, 4, 4096);
  }

  // ---- PointLSTM runner ----
  auto run_lstm = [&](const float* fea, float* fl, int C, int hd, int N,
                      const unsigned short* Wp, int KB, const float* bias,
                      unsigned short* packB){
    zero_f32_kernel<<<gblocks((long)BB * C * N), 256, 0, stream>>>(hbuf, (long)BB * C * N);
    zero_f32_kernel<<<gblocks((long)BB * hd * N), 256, 0, stream>>>(cbuf, (long)BB * hd * N);
    int cols = BB * N * KK1;
    for (int t = 0; t < TT; ++t){
      knn_kernel<<<BB, 128, 0, stream>>>(fea + (long)t * N, (long)C * TT * N, (long)TT * N, 1, N,
                                         hbuf, (long)C * N, (long)N, N,
                                         idxL, KK1, (long)N * KK1);
      pack_lstm_kernel<<<gblocks((long)BB * 2 * C * N * KK1), 256, 0, stream>>>(
          fea, hbuf, idxL, packB, C, N, KK1, t, KB);
      gather_c_kernel<<<gblocks((long)BB * hd * N * KK1), 256, 0, stream>>>(cbuf, idxL, cg, hd, N, KK1);
      launch_gemm(stream, 1, Wp, packB, KB, 4 * hd, cols, arena, nullptr, 0, bias, nullptr, nullptr);
      lstm_act_kernel<<<gblocks((long)BB * C * N), 256, 0, stream>>>(
          arena, cg, fea, hbuf, cbuf, fl, C, hd, N, KK1, t);
    }
  };

  // ---- stage (motion block) runner ----
  auto run_stage = [&](const float* fl, int C, int hd, int Nf, int nd, int k, int s,
                       const unsigned short* Wf, int KBf,
                       const float* bf, const float* gf, const float* ef,
                       const unsigned short* Wm,
                       const float* bm, const float* gm, const float* em,
                       unsigned short* packF, unsigned short* packM,
                       int Co, float* dst, int dstCtot, int chOff){
    int cols = BB * nd * k;
    for (int t = 0; t < TT; ++t){
      int tp = t ? t - 1 : 0;
      knn_kernel<<<BB, 128, 0, stream>>>(fl + (long)t * Nf,  (long)C * TT * Nf, (long)TT * Nf, s, nd,
                                         fl + (long)tp * Nf, (long)C * TT * Nf, (long)TT * Nf, Nf,
                                         idxL, k, (long)nd * k);
      pack_stage_fea_kernel<<<gblocks((long)BB * 2 * hd * nd * k), 256, 0, stream>>>(
          fl, idxL, packF, C, hd, Nf, nd, k, t, tp, s, KBf);
      pack_stage_mot_kernel<<<gblocks((long)BB * 4 * nd * k), 256, 0, stream>>>(
          fl, idxL, packM, C, Nf, nd, k, t, tp, s);
      launch_gemm(stream, 3, Wf, packF, KBf, Co, cols, arena, nullptr, 0, bf, gf, ef);
      launch_gemm(stream, 7, Wm, packM, 1,   Co, cols, arena, nullptr, 0, bm, gm, em);
      maxk_kernel<<<gblocks((long)BB * Co * nd), 256, 0, stream>>>(
          arena, dst, Co, nd, k, t, dstCtot, chOff, cols);
    }
  };

  run_lstm(fea1, fea1l, 68, 64, 128, wL1, 5, BL1, pL1);
  run_stage(fea1l, 68, 64, 128, 64, 48, 2, wS2f, 4, B2F, G2F, E2F,
            wS2m, B2M, G2M, E2M, pS2f, pS2m, 128, fea2, 132, 4);
  run_lstm(fea2, fea2l, 132, 128, 64, wL2, 9, BL2, pL2);
  run_stage(fea2l, 132, 128, 64, 32, 48, 2, wS3f, 8, B3F, G3F, E3F,
            wS3m, B3M, G3M, E3M, pS3f, pS3m, 256, fea3, 260, 4);
  run_lstm(fea3, fea3l, 260, 256, 32, wL3, 17, BL3, pL3);
  run_stage(fea3l, 260, 256, 32, 16, 24, 2, wS4f, 16, B4F, G4F, E4F,
            wS4m, B4M, G4M, E4M, pS4f, pS4m, 512, f4, 512, 0);

  // ---- stage5 + pools + head ----
  pack_plain_kernel<<<gblocks((long)BB * 512 * TT * 16), 256, 0, stream>>>(f4, pS5, 512, TT * 16, 16);
  launch_gemm(stream, 3, wS5, pS5, 16, 1024, BB * TT * 16, arena, nullptr, 0, B5, G5, E5);
  pool_stage5_kernel<<<gblocks((long)BB * 1024), 256, 0, stream>>>(arena, oe, BB * TT * 16, TT * 16);
  pool_end_kernel<<<gblocks((long)BB * 64),  256, 0, stream>>>(fea1, fea1l, oe, 68,  TT * 128, 1024, 64);
  pool_end_kernel<<<gblocks((long)BB * 128), 256, 0, stream>>>(fea2, fea2l, oe, 132, TT * 64, 1088, 128);
  pool_end_kernel<<<gblocks((long)BB * 256), 256, 0, stream>>>(fea3, fea3l, oe, 260, TT * 32, 1216, 256);
  head_kernel<<<1, 64, 0, stream>>>(oe, W6, B6, GBG, GBB, OUT);
}
